// Attention_28054726377925
// MI455X (gfx1250) — matrix operand order repaired and checked by reading, not compiled, not run on hardware
//
#include <hip/hip_runtime.h>

// ---------------------------------------------------------------------------
// Windowed attention w/ relative position bias — CDNA5 (gfx1250), wave32 WMMA.
// All GEMMs in bf16 via v_wmma_f32_16x16x32_bf16 (f32 accumulate).
//   k0: one-shot convert w_qkv -> BqkvT[384][128] bf16, w_out -> WoutT[128][128]
//   k1: qkv = x^T @ w_qkv  -> Q*scale, K (b,h,pos,dh) and V^T (b,h,dh,pos), bf16
//       (B-fragments read straight from L2-resident BqkvT; LDS only for x-tile;
//        branchless clamped gather for the x tile)
//   k2: S = Q K^T + bias(idx) ; softmax ; O = P V   (S staged in LDS,
//       branchless bias gather, global_prefetch for next K/V fragments)
//   k3: out = O @ w_out -> (b, d, 25, 25) fp32; WoutT staged in LDS via
//       global_load_async_to_lds_b128 + s_wait_asynccnt (CDNA5 async path)
// Workspace: 3*20.97MB (QKV, n padded 640) + 20.97MB (O) + 128KB (weights).
// ---------------------------------------------------------------------------

typedef __bf16 bf16;
typedef __attribute__((ext_vector_type(16))) bf16 v16bf;
typedef __attribute__((ext_vector_type(8)))  bf16 v8bf;
typedef __attribute__((ext_vector_type(4)))  bf16 v4bf;
typedef __attribute__((ext_vector_type(8)))  float v8f;

#define WIN   25
#define NTOK  625
#define NPAD  640
#define DIMD  128
#define NHEAD 4
#define DH    32
#define QSCALE 0.17677669529663687f   // 32^-0.5

static __device__ __forceinline__ v16bf mk16(v8bf lo, v8bf hi) {
  return __builtin_shufflevector(lo, hi, 0, 1, 2, 3, 4, 5, 6, 7,
                                          8, 9, 10, 11, 12, 13, 14, 15);
}
// A-fragment (16x32 bf16): lane row = lane&15, c = (lane<16?0:8),
// holds K = 2c..2c+15 contiguous (two 16B loads).
static __device__ __forceinline__ v16bf frag_a(const bf16* row, int c) {
  return mk16(*(const v8bf*)(row + 2 * c), *(const v8bf*)(row + 2 * c + 8));
}
// B-fragment (32x16 bf16): lane col = lane&15, K half kc = (lane<16?0:16),
// holds K = kc..kc+15 contiguous (32B).
static __device__ __forceinline__ v16bf frag_b(const bf16* colrow_at_kc) {
  return mk16(*(const v8bf*)(colrow_at_kc), *(const v8bf*)(colrow_at_kc + 8));
}
static __device__ __forceinline__ v8f wmma_bf16(v16bf a, v16bf b, v8f c) {
  return __builtin_amdgcn_wmma_f32_16x16x32_bf16(false, a, false, b,
                                                 (short)0, c, false, false);
}

// ---------------------------------------------------------------- kernel 0 --
// One-shot weight conversion: transposed bf16 copies (column-major over K).
__global__ __launch_bounds__(256) void wcvt_kernel(
    const float* __restrict__ w_qkv, const float* __restrict__ w_out,
    bf16* __restrict__ BqkvT, bf16* __restrict__ WoutT) {
  int g = blockIdx.x * 256 + threadIdx.x;        // 65536 threads total
  if (g < 128 * 384) {                           // w_qkv: [k=128][col=384]
    int k = g / 384, col = g - k * 384;
    BqkvT[col * 128 + k] = (bf16)w_qkv[g];
  } else {                                       // w_out: [k=128][col=128]
    int g2 = g - 128 * 384;
    int k = g2 >> 7, col = g2 & 127;
    WoutT[col * 128 + k] = (bf16)w_out[g2];
  }
}

// ---------------------------------------------------------------- kernel 1 --
__global__ __launch_bounds__(256) void qkv_kernel(
    const float* __restrict__ x, const bf16* __restrict__ BqkvT,
    bf16* __restrict__ Q, bf16* __restrict__ K, bf16* __restrict__ Vt) {
  const int mt = blockIdx.x;               // 40 tiles of 16 positions
  const int b  = blockIdx.y;               // 128 batches
  const int qbase = mt * 16;
  const int tid  = threadIdx.x;
  const int wave = tid >> 5;               // 8 waves
  const int lane = tid & 31;
  const int row  = lane & 15;
  const int ca   = (lane < 16) ? 0 : 8;
  const int kc   = (lane < 16) ? 0 : 16;

  __shared__ __align__(16) bf16 At[16][32];     // x tile, pos-major (1 KB)

  // Branchless staging setup: clamp position so loads are always in-bounds,
  // zero-select after the load (padded rows become zeros).
  const int skk = tid >> 4;                // 0..15 (k within 32-slab, +16 pair)
  const int srow = tid & 15;
  const int spos = qbase + srow;
  const bool svalid = (spos < NTOK);
  const float* xb =
      x + (size_t)b * DIMD * NTOK + (svalid ? spos : (NTOK - 1));

  v8f acc[3] = {};
  for (int kb = 0; kb < DIMD; kb += 32) {
    float v0 = xb[(size_t)(kb + skk) * NTOK];        // unconditional loads
    float v1 = xb[(size_t)(kb + skk + 16) * NTOK];
    At[srow][skk]      = (bf16)(svalid ? v0 : 0.0f);
    At[srow][skk + 16] = (bf16)(svalid ? v1 : 0.0f);
    __syncthreads();
    v16bf a = frag_a(&At[row][0], ca);
#pragma unroll
    for (int nt = 0; nt < 3; ++nt) {
      int ct = wave * 3 + nt;                       // 24 col tiles
      v16bf bb = frag_b(BqkvT + (size_t)(ct * 16 + row) * 128 + kb + kc);
      acc[nt] = wmma_bf16(a, bb, acc[nt]);
    }
    __syncthreads();
  }
  // Store: cols 0..127 = Q (pre-scaled), 128..255 = K, 256..383 = V^T
#pragma unroll
  for (int nt = 0; nt < 3; ++nt) {
    int col  = (wave * 3 + nt) * 16 + row;
    int comp = col >> 7;
    int dcol = col & 127;
    int head = dcol >> 5;
    int dhi  = dcol & 31;
    size_t bh = (size_t)b * NHEAD + head;
#pragma unroll
    for (int r = 0; r < 8; ++r) {
      int pos = qbase + r + ((lane >> 4) << 3);     // padded rows store zeros
      float v = acc[nt][r];
      if (comp == 0)      Q[(bh * NPAD + pos) * DH + dhi] = (bf16)(v * QSCALE);
      else if (comp == 1) K[(bh * NPAD + pos) * DH + dhi] = (bf16)v;
      else                Vt[(bh * DH + dhi) * NPAD + pos] = (bf16)v;
    }
  }
}

// ---------------------------------------------------------------- kernel 2 --
__global__ __launch_bounds__(64) void attn_kernel(
    const bf16* __restrict__ Q, const bf16* __restrict__ K,
    const bf16* __restrict__ Vt, const float* __restrict__ bias_table,
    bf16* __restrict__ O) {
  const int mt = blockIdx.x;               // 40 query tiles
  const int h  = blockIdx.y;               // 4 heads
  const int b  = blockIdx.z;               // 128 batches
  const int qbase = mt * 16;
  const int tid  = threadIdx.x;
  const int wave = tid >> 5;               // 2 waves
  const int lane = tid & 31;
  const int row  = lane & 15;
  const int hi8  = (lane >> 4) << 3;
  const int ca   = (lane < 16) ? 0 : 8;
  const int kc   = (lane < 16) ? 0 : 16;

  __shared__ __align__(16) float Ssm[16][NPAD];   // 40 KB
  __shared__ __align__(16) bf16  Psm[16][NPAD];   // 20 KB
  __shared__ float redmax[16][4];
  __shared__ float redsum[16][4];

  const size_t bh = (size_t)b * NHEAD + h;
  const bf16* Qb = Q  + bh * NPAD * DH;
  const bf16* Kb = K  + bh * NPAD * DH;
  const bf16* Vb = Vt + bh * DH * NPAD;

  // ---- phase 1: S = Q K^T + bias, masked, staged in LDS -------------------
  {
    v16bf a = frag_a(Qb + (size_t)(qbase + row) * DH, ca);   // dh==K==32
    for (int j = wave; j < NPAD / 16; j += 2) {
      int kbase = j * 16;
      // prefetch next K tile fragment for this wave (global_prefetch_b8)
      int jn = (j + 2 < NPAD / 16) ? j + 2 : j;
      __builtin_prefetch(Kb + (size_t)(jn * 16 + row) * DH + kc, 0, 3);

      v16bf bb = frag_b(Kb + (size_t)(kbase + row) * DH + kc);
      v8f s = {};
      s = wmma_bf16(a, bb, s);

      int kpos = kbase + row;                   // this lane's key column
      int ki = kpos / WIN, kj = kpos - ki * WIN;
      bool kvalid = (kpos < NTOK);
      // Branchless gather: clamp invalid indices to 0, load unconditionally
      // so all 8 loads pipeline under a single wait.
      float bias[8];
#pragma unroll
      for (int r = 0; r < 8; ++r) {
        int qpos = qbase + hi8 + r;
        int qi = qpos / WIN, qj = qpos - qi * WIN;
        int idx = (qi - ki + WIN - 1) * (2 * WIN - 1) + (qj - kj + WIN - 1);
        bool valid = kvalid && (qpos < NTOK);
        idx = valid ? idx : 0;
        bias[r] = bias_table[idx * NHEAD + h];
      }
#pragma unroll
      for (int r = 0; r < 8; ++r) {
        int m = r + hi8;
        int qpos = qbase + m;
        float val = !kvalid ? -1e30f
                  : (qpos >= NTOK ? 0.0f : s[r] + bias[r]);
        Ssm[m][kpos] = val;
      }
    }
  }
  __syncthreads();

  // ---- phase 2: two-pass row softmax (unnormalized P in bf16) -------------
  {
    const int srow = tid & 15, slot = tid >> 4;  // 4 threads per row
    const float4* sp = (const float4*)&Ssm[srow][slot * 160];
    float mx = -3e38f;
#pragma unroll 4
    for (int i = 0; i < 40; ++i) {
      float4 v = sp[i];
      mx = fmaxf(mx, fmaxf(fmaxf(v.x, v.y), fmaxf(v.z, v.w)));
    }
    redmax[srow][slot] = mx;
    __syncthreads();
    float rmax = fmaxf(fmaxf(redmax[srow][0], redmax[srow][1]),
                       fmaxf(redmax[srow][2], redmax[srow][3]));
    float sum = 0.f;
    v4bf* pp = (v4bf*)&Psm[srow][slot * 160];
#pragma unroll 4
    for (int i = 0; i < 40; ++i) {
      float4 v = sp[i];
      float e0 = __expf(v.x - rmax), e1 = __expf(v.y - rmax);
      float e2 = __expf(v.z - rmax), e3 = __expf(v.w - rmax);
      v4bf pk = {(bf16)e0, (bf16)e1, (bf16)e2, (bf16)e3};
      pp[i] = pk;
      sum += (e0 + e1) + (e2 + e3);
    }
    redsum[srow][slot] = sum;
  }
  __syncthreads();

  // ---- phase 3: O = P V (wave owns one 16-wide dh half) -------------------
  v8f o = {};
  const int dhcol = wave * 16 + row;
  const bf16* vrow = Vb + (size_t)dhcol * NPAD;
  for (int kb = 0; kb < NPAD; kb += 32) {
    int kn = (kb + 32 < NPAD) ? kb + 32 : kb;
    __builtin_prefetch(vrow + kn + kc, 0, 3);
    v16bf a  = frag_a(&Psm[row][kb], ca);
    v16bf bb = frag_b(vrow + kb + kc);
    o = wmma_bf16(a, bb, o);
  }
#pragma unroll
  for (int r = 0; r < 8; ++r) {
    int m = r + hi8;
    float l = redsum[m][0] + redsum[m][1] + redsum[m][2] + redsum[m][3];
    int pos = qbase + m;
    O[((size_t)b * NPAD + pos) * DIMD + h * DH + dhcol] = (bf16)(o[r] / l);
  }
}

// ---------------------------------------------------------------- kernel 3 --
__global__ __launch_bounds__(128) void proj_kernel(
    const bf16* __restrict__ O, const bf16* __restrict__ WoutT,
    float* __restrict__ out) {
  const int mt = blockIdx.x;               // 40
  const int b  = blockIdx.y;               // 128
  const int qbase = mt * 16;
  const int tid  = threadIdx.x;
  const int wave = tid >> 5;               // 4 waves
  const int lane = tid & 31;
  const int row  = lane & 15;
  const int hi8  = (lane >> 4) << 3;
  const int ca   = (lane < 16) ? 0 : 8;
  const int kc   = (lane < 16) ? 0 : 16;

  // Stage WoutT (already bf16, already transposed) into LDS with the CDNA5
  // async copy path: 2048 x 16B chunks, ASYNCcnt-tracked, then barrier.
  __shared__ __align__(16) bf16 Wt[128][128];   // 32 KB
  {
    unsigned lds_base = (unsigned)(size_t)&Wt[0][0];  // low 32 bits = LDS offset
    unsigned long long gbase = (unsigned long long)WoutT;
#pragma unroll
    for (int i = 0; i < 16; ++i) {
      unsigned byte_off = (unsigned)(tid + i * 128) * 16u;
      asm volatile("global_load_async_to_lds_b128 %0, %1, %2"
                   :: "v"(lds_base + byte_off), "v"(byte_off), "s"(gbase)
                   : "memory");
    }
    asm volatile("s_wait_asynccnt 0x0" ::: "memory");
  }
  __syncthreads();

  v8f acc[2] = {};
  const bf16* Ob = O + ((size_t)b * NPAD + qbase + row) * DIMD;
#pragma unroll
  for (int kb = 0; kb < DIMD; kb += 32) {
    v16bf a = frag_a(Ob + kb, ca);
#pragma unroll
    for (int nt = 0; nt < 2; ++nt) {
      int ct = wave * 2 + nt;                   // 8 col tiles
      v16bf bb = frag_b(&Wt[ct * 16 + row][kb + kc]);
      acc[nt] = wmma_bf16(a, bb, acc[nt]);
    }
  }
#pragma unroll
  for (int nt = 0; nt < 2; ++nt) {
    int col = (wave * 2 + nt) * 16 + row;
#pragma unroll
    for (int r = 0; r < 8; ++r) {
      int pos = qbase + r + hi8;
      if (pos < NTOK)
        out[((size_t)b * DIMD + col) * NTOK + pos] = acc[nt][r];
    }
  }
}

// ---------------------------------------------------------------- launcher --
extern "C" void kernel_launch(void* const* d_in, const int* in_sizes, int n_in,
                              void* d_out, int out_size, void* d_ws,
                              size_t ws_size, hipStream_t stream) {
  const float* x          = (const float*)d_in[0];
  const float* w_qkv      = (const float*)d_in[1];
  const float* w_out      = (const float*)d_in[2];
  const float* bias_table = (const float*)d_in[3];

  char* ws = (char*)d_ws;
  const size_t qkv_bytes = (size_t)128 * NHEAD * NPAD * DH * sizeof(bf16); // 20.97MB
  bf16* Q     = (bf16*)(ws);
  bf16* K     = (bf16*)(ws + qkv_bytes);
  bf16* Vt    = (bf16*)(ws + 2 * qkv_bytes);
  bf16* Obuf  = (bf16*)(ws + 3 * qkv_bytes);     // (b, pos_pad, 128) bf16
  bf16* BqkvT = (bf16*)(ws + 4 * qkv_bytes);     // 384x128 bf16 (96 KB)
  bf16* WoutT = BqkvT + 384 * 128;               // 128x128 bf16 (32 KB)

  wcvt_kernel<<<dim3(256), 256, 0, stream>>>(w_qkv, w_out, BqkvT, WoutT);
  qkv_kernel<<<dim3(40, 128), 256, 0, stream>>>(x, BqkvT, Q, K, Vt);
  attn_kernel<<<dim3(40, NHEAD, 128), 64, 0, stream>>>(Q, K, Vt, bias_table,
                                                       Obuf);
  proj_kernel<<<dim3(40, 128), 128, 0, stream>>>(Obuf, WoutT, (float*)d_out);
}
